// HAN_67585605370059
// MI455X (gfx1250) — compile-verified, matrix operand order checked by reading, not run on hardware
//
#include <hip/hip_runtime.h>

// ---------------------------------------------------------------------------
// HAN forward for MI455X (gfx1250, wave32, WMMA).
// Dense feature transforms run on v_wmma_f32_16x16x32_f16 with double-buffered
// B fragments; A tiles are staged to LDS via TDM tensor_load_to_lds (or async
// global->LDS loads) when the toolchain exposes the builtins. Edge softmax /
// aggregation is the bandwidth-bound part and uses float4 gathers + f32
// global atomics.
// ---------------------------------------------------------------------------

#define kN   50000
#define kE   800000
#define kM   3
#define kH   4
#define kD   64
#define kP   256
#define kIN  256
#define kHid 128
#define kSlope 0.2f

typedef __attribute__((ext_vector_type(16))) _Float16 v16h;
typedef __attribute__((ext_vector_type(8)))  _Float16 v8h;
typedef __attribute__((ext_vector_type(8)))  float    v8f;
typedef unsigned int v4u __attribute__((ext_vector_type(4)));
typedef int          v4i __attribute__((ext_vector_type(4)));
typedef int          v8i __attribute__((ext_vector_type(8)));

// --- CDNA5 data-mover capability tiers (guarded: absent builtin == tier off)
#if __has_builtin(__builtin_amdgcn_tensor_load_to_lds) && \
    __has_builtin(__builtin_amdgcn_s_wait_tensorcnt)
#define HAN_TDM 1
#else
#define HAN_TDM 0
#endif
#if __has_builtin(__builtin_amdgcn_global_load_async_to_lds_b128) && \
    __has_builtin(__builtin_amdgcn_s_wait_asynccnt)
#define HAN_ASYNC 1
#else
#define HAN_ASYNC 0
#endif

// ---- small device helpers -------------------------------------------------

static __device__ __forceinline__ unsigned f2ord(float f) {
  unsigned u = __float_as_uint(f);
  return (u & 0x80000000u) ? ~u : (u | 0x80000000u);
}
static __device__ __forceinline__ float ord2f(unsigned u) {
  return __uint_as_float((u & 0x80000000u) ? (u & 0x7FFFFFFFu) : ~u);
}
static __device__ __forceinline__ unsigned hashu(unsigned a, unsigned b) {
  unsigned x = a * 0x9E3779B9u ^ b * 0x85EBCA6Bu;
  x ^= x >> 16; x *= 0x7FEB352Du;
  x ^= x >> 15; x *= 0x846CA68Bu;
  x ^= x >> 16;
  return x;
}
static __device__ __forceinline__ float u01f(unsigned h) {
  return ((h >> 8) + 0.5f) * (1.0f / 16777216.0f);
}
static __device__ __forceinline__ float gaussf(unsigned ha, unsigned hb) {
  float u1 = u01f(ha), u2 = u01f(hb);
  return sqrtf(-2.0f * logf(u1)) * cosf(6.28318530718f * u2);
}

// ---- elementwise utility kernels ------------------------------------------

__global__ __launch_bounds__(256) void han_f2h(const float* __restrict__ in,
                                               _Float16* __restrict__ out, long n) {
  long i = (long)blockIdx.x * blockDim.x + threadIdx.x;
  if (i < n) out[i] = (_Float16)in[i];
}

__global__ __launch_bounds__(256) void han_zero(float* __restrict__ p, long n) {
  long i = (long)blockIdx.x * blockDim.x + threadIdx.x;
  if (i < n) p[i] = 0.0f;
}

__global__ __launch_bounds__(256) void han_copy(const float* __restrict__ a,
                                                float* __restrict__ b, long n) {
  long i = (long)blockIdx.x * blockDim.x + threadIdx.x;
  if (i < n) b[i] = a[i];
}

__global__ __launch_bounds__(256) void han_relu(float* __restrict__ p, long n) {
  long i = (long)blockIdx.x * blockDim.x + threadIdx.x;
  if (i < n) p[i] = fmaxf(p[i], 0.0f);
}

// W [K, Ncol] f32 row-major -> Wt [Ncol, K] f16 (so WMMA B-fragment loads are
// contiguous, same addressing pattern as the A-fragment).
__global__ __launch_bounds__(256) void han_transpose_f2h(const float* __restrict__ Wsrc,
                                                         _Float16* __restrict__ Wt,
                                                         int K, int Ncol) {
  int i = blockIdx.x * 256 + threadIdx.x;
  if (i < K * Ncol) {
    int k = i / Ncol;
    int n = i - k * Ncol;
    Wt[(size_t)n * K + k] = (_Float16)Wsrc[i];
  }
}

// ---- WMMA GEMM: C[rows, ncols] = A[rows,256] * B[256, ncols] ---------------
// A: f16 row-major. Bt: f16, column-of-B-major ([ncols][256]).
// Block = 256 threads = 8 waves; block owns 16 rows (A tile staged in LDS via
// TDM / async-LDS loads when available), each wave produces 16x16 output
// tiles, K swept 256 in 8 WMMA steps with register double-buffered B.

__global__ __launch_bounds__(256) void han_gemm_wmma(const _Float16* __restrict__ A,
                                                     const _Float16* __restrict__ Bt,
                                                     float* __restrict__ C, int ncols) {
  __shared__ __align__(16) _Float16 ldsA[16 * 256];

  const int block_row = blockIdx.x * 16;

#if HAN_TDM
  // Tensor Data Mover: DMA the 16x512B A tile straight into LDS.
  if ((threadIdx.x >> 5) == 0) {
    unsigned long long ga =
        (unsigned long long)A + (unsigned long long)block_row * 512ull;
    unsigned lds_base =
        (unsigned)(unsigned long long)(__attribute__((address_space(3))) char*)&ldsA[0];
    v4u g0;
    g0.x = 1u;                                            // count=1 (valid D#)
    g0.y = lds_base;                                      // lds_addr
    g0.z = (unsigned)(ga & 0xFFFFFFFFull);                // global_addr[31:0]
    g0.w = (unsigned)((ga >> 32) & 0x01FFFFFFull) | (2u << 30); // addr hi | type=2
    // D# group 1: data_size=8B, tensor_dim0=64, tensor_dim1=big bound,
    // tile 64x16, tensor_dim0_stride=64 (contiguous rows).
    v8i g1;
    g1[0] = (int)(3u << 16);              // [17:16] data_size = 3 (8 bytes)
    g1[1] = (int)(64u << 16);             // [31:16] tensor_dim0[15:0] = 64
    g1[2] = (int)(0xFFFFu << 16);         // [31:16] tensor_dim1[15:0] = 0xFFFF
    g1[3] = (int)((64u << 16) | 0x0FFFu); // tile_dim0=64 | tensor_dim1[31:16]
    g1[4] = 16;                           // tile_dim1=16, tile_dim2=0
    g1[5] = 64;                           // tensor_dim0_stride[31:0] = 64
    g1[6] = 0;
    g1[7] = 0;
    v4i zz4 = (v4i){0, 0, 0, 0};
    v8i zz8 = (v8i){0, 0, 0, 0, 0, 0, 0, 0};
    __builtin_amdgcn_tensor_load_to_lds(g0, g1, zz4, zz4, zz8, 0);
    __builtin_amdgcn_s_wait_tensorcnt(0);
  }
  __syncthreads();
#elif HAN_ASYNC
  {
    const uint4* srcg = (const uint4*)(A + (size_t)block_row * 256);
    uint4* dstl = (uint4*)ldsA;
    __builtin_amdgcn_global_load_async_to_lds_b128(
        (__attribute__((address_space(1))) void*)(srcg + threadIdx.x),
        (__attribute__((address_space(3))) void*)(dstl + threadIdx.x), 0, 0);
    __builtin_amdgcn_global_load_async_to_lds_b128(
        (__attribute__((address_space(1))) void*)(srcg + threadIdx.x + 256),
        (__attribute__((address_space(3))) void*)(dstl + threadIdx.x + 256), 0, 0);
    __builtin_amdgcn_s_wait_asynccnt(0);
  }
  __syncthreads();
#else
  {
    // 16 rows x 256 halves = 8 KB = 512 x b128; 2 loads per thread.
    const uint4* src = (const uint4*)(A + (size_t)block_row * 256);
    uint4* dst = (uint4*)ldsA;
    dst[threadIdx.x]       = src[threadIdx.x];
    dst[threadIdx.x + 256] = src[threadIdx.x + 256];
  }
  __syncthreads();
#endif

  const int wave  = threadIdx.x >> 5;         // wave32
  const int lane  = threadIdx.x & 31;
  const int arow  = lane & 15;                // A-fragment row for this lane
  const int khalf = (lane >> 4) << 3;         // 0 or 8: K sub-offset per layout
  const int nct   = ncols >> 4;

  for (int ct = wave; ct < nct; ct += 8) {
    const int colb = (ct << 4) + (lane & 15);
    const _Float16* bp = Bt + (size_t)colb * 256 + khalf;
    if (ct + 8 < nct)
      __builtin_prefetch(Bt + (size_t)(colb + 128) * 256, 0, 1);

    // register double-buffer for the B fragment: fetch k+32 before using k
    v8h blo = *(const v8h*)(bp);
    v8h bhi = *(const v8h*)(bp + 16);
    v8f acc = {};
#pragma unroll
    for (int k = 0; k < 256; k += 32) {
      const v8h clo = blo, chi = bhi;
      if (k + 32 < 256) {
        blo = *(const v8h*)(bp + k + 32);
        bhi = *(const v8h*)(bp + k + 48);
      }
      // A fragment: lane holds K = k0..k0+7 and k0+16..k0+23 of its row.
      const v8h alo = *(const v8h*)(&ldsA[arow * 256 + k + khalf]);
      const v8h ahi = *(const v8h*)(&ldsA[arow * 256 + k + khalf + 16]);
      const v16h a = __builtin_shufflevector(alo, ahi,
          0, 1, 2, 3, 4, 5, 6, 7, 8, 9, 10, 11, 12, 13, 14, 15);
      const v16h b = __builtin_shufflevector(clo, chi,
          0, 1, 2, 3, 4, 5, 6, 7, 8, 9, 10, 11, 12, 13, 14, 15);
      acc = __builtin_amdgcn_wmma_f32_16x16x32_f16(
          false, a, false, b, (short)0, acc, false, false);
    }
    // C/D layout: VGPR r -> M=r (lanes 0-15) / M=r+8 (lanes 16-31), N=lane&15.
    const int rbase = (lane >> 4) << 3;
    float* crow = C + (size_t)(block_row + rbase) * ncols + (ct << 4) + (lane & 15);
#pragma unroll
    for (int r = 0; r < 8; ++r)
      crow[(size_t)r * ncols] = acc[r];
  }
}

// ---- GAT pieces ------------------------------------------------------------

__global__ __launch_bounds__(256) void han_el_er(const float* __restrict__ feat,
                                                 const float* __restrict__ al,
                                                 const float* __restrict__ ar,
                                                 float* __restrict__ el,
                                                 float* __restrict__ er) {
  int i = blockIdx.x * 256 + threadIdx.x;   // over N*H
  if (i >= kN * kH) return;
  int n = i >> 2, h = i & 3;
  const float* f = feat + (size_t)n * kP + h * kD;
  const float* a = al + h * kD;
  const float* b = ar + h * kD;
  float sl = 0.0f, sr = 0.0f;
#pragma unroll 4
  for (int d = 0; d < kD; ++d) {
    float fv = f[d];
    sl += fv * a[d];
    sr += fv * b[d];
  }
  el[i] = sl;
  er[i] = sr;
}

__global__ __launch_bounds__(256) void han_init_maxden(unsigned* __restrict__ maxb,
                                                       float* __restrict__ den) {
  int i = blockIdx.x * 256 + threadIdx.x;
  if (i < kN * kH) { maxb[i] = 0u; den[i] = 0.0f; }   // 0u == ord(-inf) floor
}

__global__ __launch_bounds__(256) void han_edge1(const int* __restrict__ src,
                                                 const int* __restrict__ dst,
                                                 const float* __restrict__ el,
                                                 const float* __restrict__ er,
                                                 float* __restrict__ ebuf,
                                                 unsigned* __restrict__ maxb) {
  int i = blockIdx.x * 256 + threadIdx.x;   // over E*H
  if (i >= kE * kH) return;
  int e = i >> 2, h = i & 3;
  int s = src[e], d = dst[e];
  float v = el[s * kH + h] + er[d * kH + h];
  v = (v > 0.0f) ? v : v * kSlope;          // leaky relu
  ebuf[i] = v;
  atomicMax(&maxb[d * kH + h], f2ord(v));   // ordered-uint float max
}

__global__ __launch_bounds__(256) void han_edge2(const int* __restrict__ dst,
                                                 float* __restrict__ ebuf,
                                                 const unsigned* __restrict__ maxb,
                                                 float* __restrict__ den) {
  int i = blockIdx.x * 256 + threadIdx.x;
  if (i >= kE * kH) return;
  int e = i >> 2, h = i & 3;
  int d = dst[e];
  float ex = __expf(ebuf[i] - ord2f(maxb[d * kH + h]));
  ebuf[i] = ex;                              // reuse buffer as exp(e - max)
  atomicAdd(&den[d * kH + h], ex);
}

// alpha-weighted scatter: 64 threads/edge, float4 gather + 4 f32 atomics.
__global__ __launch_bounds__(256) void han_edge3(const int* __restrict__ src,
                                                 const int* __restrict__ dst,
                                                 const float* __restrict__ exbuf,
                                                 const float* __restrict__ den,
                                                 const float* __restrict__ feat,
                                                 float* __restrict__ z, int m) {
  long t = (long)blockIdx.x * 256 + threadIdx.x;
  if (t >= (long)kE * 64) return;
  int e = (int)(t >> 6);
  int q = ((int)t & 63) << 2;                // feature offset 0..252
  int h = q >> 6;
  int s = src[e], d = dst[e];
  float alpha = exbuf[e * kH + h] / den[d * kH + h];
  const float4 f = *(const float4*)(feat + (size_t)s * kP + q);
  float* zp = z + ((size_t)d * kM + m) * kP + q;
  atomicAdd(zp + 0, alpha * f.x);
  atomicAdd(zp + 1, alpha * f.y);
  atomicAdd(zp + 2, alpha * f.z);
  atomicAdd(zp + 3, alpha * f.w);
}

// ---- semantic attention ----------------------------------------------------

__global__ __launch_bounds__(256) void han_sem_score(const float* __restrict__ proj,
                                                     const float* __restrict__ b1,
                                                     const float* __restrict__ W2,
                                                     float* __restrict__ wsum) {
  __shared__ float part[kM];
  if (threadIdx.x < kM) part[threadIdx.x] = 0.0f;
  __syncthreads();
  int row = blockIdx.x * 256 + threadIdx.x;   // over N*M (z flattened [n][m])
  if (row < kN * kM) {
    int m = row % kM;
    const float* p = proj + (size_t)row * kHid;
    float s = 0.0f;
#pragma unroll 4
    for (int j = 0; j < kHid; ++j) s += tanhf(p[j] + b1[j]) * W2[j];
    atomicAdd(&part[m], s);
  }
  __syncthreads();
  if (threadIdx.x < kM) atomicAdd(&wsum[threadIdx.x], part[threadIdx.x]);
}

__global__ void han_beta(const float* __restrict__ wsum,
                         float* __restrict__ beta, float* att_out) {
  if (blockIdx.x == 0 && threadIdx.x == 0) {
    float w0 = wsum[0] / (float)kN;
    float w1 = wsum[1] / (float)kN;
    float w2 = wsum[2] / (float)kN;
    float mx = fmaxf(w0, fmaxf(w1, w2));
    float e0 = __expf(w0 - mx), e1 = __expf(w1 - mx), e2 = __expf(w2 - mx);
    float inv = 1.0f / (e0 + e1 + e2);
    beta[0] = e0 * inv; beta[1] = e1 * inv; beta[2] = e2 * inv;
    if (att_out) { att_out[0] = beta[0]; att_out[1] = beta[1]; att_out[2] = beta[2]; }
  }
}

__global__ __launch_bounds__(256) void han_combine(const float* __restrict__ z,
                                                   const float* __restrict__ beta,
                                                   float* __restrict__ out) {
  long i = (long)blockIdx.x * 256 + threadIdx.x;  // over N*P
  if (i >= (long)kN * kP) return;
  long n = i >> 8;
  int  p = (int)i & 255;
  const float* zr = z + (size_t)n * kM * kP + p;
  out[i] = beta[0] * zr[0] + beta[1] * zr[kP] + beta[2] * zr[2 * kP];
}

// ---- post-processing -------------------------------------------------------

__global__ __launch_bounds__(256) void han_l2norm(float* __restrict__ v) {
  int row  = blockIdx.x * 8 + (threadIdx.x >> 5);  // one wave32 per row
  int lane = threadIdx.x & 31;
  if (row >= kN) return;
  float* r = v + (size_t)row * kP;
  float s = 0.0f;
  for (int j = lane; j < kP; j += 32) { float x = r[j]; s += x * x; }
#pragma unroll
  for (int off = 16; off > 0; off >>= 1) s += __shfl_xor(s, off, 32);
  float inv = 1.0f / fmaxf(sqrtf(s), 1e-12f);
  for (int j = lane; j < kP; j += 32) r[j] *= inv;
}

__global__ __launch_bounds__(256) void han_reparam(const float* __restrict__ mean,
                                                   const float* __restrict__ lsd,
                                                   float* __restrict__ z1,
                                                   float* __restrict__ z2, long n) {
  long i = (long)blockIdx.x * 256 + threadIdx.x;
  if (i >= n) return;
  float sd = __expf(lsd[i]);
  float mn = mean[i];
  unsigned ui = (unsigned)i;
  z1[i] = mn + gaussf(hashu(1u, ui), hashu(2u, ui)) * sd;
  z2[i] = mn + gaussf(hashu(3u, ui), hashu(4u, ui)) * sd;
}

__global__ __launch_bounds__(256) void han_negs(const float* __restrict__ h1,
                                                const float* __restrict__ mean,
                                                const float* __restrict__ lsd,
                                                const int* __restrict__ epoch_p,
                                                float* __restrict__ negs, long n) {
  long i = (long)blockIdx.x * 256 + threadIdx.x;
  if (i >= n) return;
  int ep  = epoch_p[0];
  int vae = (int)(0.4f * (float)ep);          // NEG_NUM*2/25 * epoch
  if (vae < 0) vae = 0;
  if (vae > 10) vae = 10;
  int enc = 10 - vae;
  float hv  = h1[i];
  float mn2 = mean[i] * 2.0f;                 // DEVI_NUM
  float sd  = __expf(lsd[i]);
  unsigned ui = (unsigned)i;
#pragma unroll 1
  for (int k = 0; k < 10; ++k) {
    float v;
    if (k < enc) {
      float u = u01f(hashu(100u + (unsigned)k, ui));
      v = (u < 0.5f) ? hv * 2.0f : 0.0f;      // dropout mask / keep
    } else {
      int g = k - enc;
      v = mn2 + gaussf(hashu(200u + (unsigned)g, ui),
                       hashu(300u + (unsigned)g, ui)) * sd;
    }
    negs[(size_t)k * n + i] = v;
  }
}

// ---------------------------------------------------------------------------

extern "C" void kernel_launch(void* const* d_in, const int* in_sizes, int n_in,
                              void* d_out, int out_size, void* d_ws, size_t ws_size,
                              hipStream_t stream) {
  (void)in_sizes; (void)n_in; (void)out_size; (void)ws_size;

  const float* x   = (const float*)d_in[0];
  const float* W   = (const float*)d_in[1];   // [4,3,256,256]
  const float* al  = (const float*)d_in[2];   // [4,3,4,64]
  const float* ar  = (const float*)d_in[3];
  const float* sW1 = (const float*)d_in[4];   // [4,256,128]
  const float* sb1 = (const float*)d_in[5];   // [4,128]
  const float* sW2 = (const float*)d_in[6];   // [4,128]
  const int*   ei  = (const int*)d_in[7];     // [3,2,E]
  const int*   ep  = (const int*)d_in[8];     // [1]

  float* out = (float*)d_out;
  const size_t NP = (size_t)kN * kP;
  float* o_h1   = out;
  float* o_h2   = o_h1 + NP;
  float* o_att  = o_h2 + NP;        // 3
  float* o_z1   = o_att + 3;
  float* o_z2   = o_z1 + NP;
  float* o_mean = o_z2 + NP;
  float* o_lsd  = o_mean + NP;
  float* o_negs = o_lsd + NP;       // 10*NP

  // --- workspace carve-out ---
  char* wsp = (char*)d_ws;
  size_t off = 0;
  auto walloc = [&](size_t bytes) -> void* {
    void* p = wsp + off;
    off = (off + bytes + 255) & ~(size_t)255;
    return p;
  };
  _Float16* hf16  = (_Float16*)walloc((size_t)kN * kIN * 2);
  _Float16* wt16  = (_Float16*)walloc((size_t)kIN * kP * 2);
  _Float16* swt16 = (_Float16*)walloc((size_t)kP * kHid * 2);
  float*    feat  = (float*)walloc((size_t)kN * kP * 4);
  float*    el    = (float*)walloc((size_t)kN * kH * 4);
  float*    er    = (float*)walloc((size_t)kN * kH * 4);
  unsigned* maxb  = (unsigned*)walloc((size_t)kN * kH * 4);
  float*    den   = (float*)walloc((size_t)kN * kH * 4);
  float*    ebuf  = (float*)walloc((size_t)kE * kH * 4);
  float*    z     = (float*)walloc((size_t)kN * kM * kP * 4);
  _Float16* zf16  = (_Float16*)walloc((size_t)kN * kM * kP * 2);
  float*    proj  = (float*)walloc((size_t)kN * kM * kHid * 4);
  float*    small = (float*)walloc(64);
  float*    h1a   = (float*)walloc((size_t)kN * kP * 4);
  float* wsum = small;
  float* beta = small + 8;

  auto run_layer = [&](int layer, const float* h_in, float* h_out,
                       float* att_out, bool do_relu) {
    const long nIn = (long)kN * kIN;
    const long nZ  = (long)kN * kM * kP;
    han_f2h<<<dim3((unsigned)((nIn + 255) / 256)), 256, 0, stream>>>(h_in, hf16, nIn);
    han_zero<<<dim3((unsigned)((nZ + 255) / 256)), 256, 0, stream>>>(z, nZ);
    for (int m = 0; m < kM; ++m) {
      const float* Wm = W + ((size_t)layer * kM + m) * kIN * kP;
      han_transpose_f2h<<<dim3((kIN * kP + 255) / 256), 256, 0, stream>>>(Wm, wt16, kIN, kP);
      han_gemm_wmma<<<dim3(kN / 16), 256, 0, stream>>>(hf16, wt16, feat, kP);
      const float* alm = al + ((size_t)layer * kM + m) * kH * kD;
      const float* arm = ar + ((size_t)layer * kM + m) * kH * kD;
      han_el_er<<<dim3((kN * kH + 255) / 256), 256, 0, stream>>>(feat, alm, arm, el, er);
      han_init_maxden<<<dim3((kN * kH + 255) / 256), 256, 0, stream>>>(maxb, den);
      const int* srcp = ei + (size_t)m * 2 * kE;
      const int* dstp = srcp + kE;
      han_edge1<<<dim3((kE * kH + 255) / 256), 256, 0, stream>>>(srcp, dstp, el, er, ebuf, maxb);
      han_edge2<<<dim3((kE * kH + 255) / 256), 256, 0, stream>>>(dstp, ebuf, maxb, den);
      han_edge3<<<dim3((unsigned)(((long)kE * 64 + 255) / 256)), 256, 0, stream>>>(
          srcp, dstp, ebuf, den, feat, z, m);
    }
    if (do_relu)
      han_relu<<<dim3((unsigned)((nZ + 255) / 256)), 256, 0, stream>>>(z, nZ);
    // semantic attention
    han_f2h<<<dim3((unsigned)((nZ + 255) / 256)), 256, 0, stream>>>(z, zf16, nZ);
    han_transpose_f2h<<<dim3((kP * kHid + 255) / 256), 256, 0, stream>>>(
        sW1 + (size_t)layer * kP * kHid, swt16, kP, kHid);
    han_gemm_wmma<<<dim3(kN * kM / 16), 256, 0, stream>>>(zf16, swt16, proj, kHid);
    han_zero<<<dim3(1), 32, 0, stream>>>(wsum, 16);
    han_sem_score<<<dim3((kN * kM + 255) / 256), 256, 0, stream>>>(
        proj, sb1 + (size_t)layer * kHid, sW2 + (size_t)layer * kHid, wsum);
    han_beta<<<dim3(1), 32, 0, stream>>>(wsum, beta, att_out);
    han_combine<<<dim3((unsigned)((NP + 255) / 256)), 256, 0, stream>>>(z, beta, h_out);
  };

  run_layer(0, x, h1a, nullptr, true);          // relu
  run_layer(1, h1a, o_h1, o_att, false);        // h1 + att_mp
  han_copy<<<dim3((unsigned)((NP + 255) / 256)), 256, 0, stream>>>(o_h1, o_h2, (long)NP);
  run_layer(2, o_h1, o_mean, nullptr, false);   // mean (raw)
  run_layer(3, o_h1, o_lsd, nullptr, false);    // log_std (raw)
  han_l2norm<<<dim3(kN / 8), 256, 0, stream>>>(o_mean);
  han_l2norm<<<dim3(kN / 8), 256, 0, stream>>>(o_lsd);
  han_reparam<<<dim3((unsigned)((NP + 255) / 256)), 256, 0, stream>>>(
      o_mean, o_lsd, o_z1, o_z2, (long)NP);
  han_negs<<<dim3((unsigned)((NP + 255) / 256)), 256, 0, stream>>>(
      o_h1, o_mean, o_lsd, ep, o_negs, (long)NP);
}
